// GAT_GCN_33904471834792
// MI455X (gfx1250) — matrix operand order repaired and checked by reading, not compiled, not run on hardware
//
#include <hip/hip_runtime.h>

#define NN    20000
#define NE    160000
#define E2    180000   // NE + NN self loops
#define NH    10
#define FX    78
#define HF    780      // NH*FX
#define NG    512
#define SEQL  1000
#define SEQP  1024     // padded sequence length
#define VOC   26
#define EMBD  128
#define NFILT 32
#define CO    121
#define CFL   3872     // NFILT*CO
#define LDSS  40       // padded LDS row stride (bf16 elems), 80B rows, 16B aligned
#define WPLD  1024     // padded K stride of packed conv weights

typedef __bf16 bf16_t;
typedef bf16_t v16bf __attribute__((ext_vector_type(16)));
typedef float  v8f   __attribute__((ext_vector_type(8)));

struct alignas(16) U128 { unsigned int w[4]; };
struct alignas(8)  U64  { unsigned int x, y; };
union FragU { v16bf v; U128 q[2]; };

__device__ inline bf16_t f2bf(float f) {
  union { float f; unsigned u; } a; a.f = f;
  unsigned r = a.u + 0x7FFFu + ((a.u >> 16) & 1u);   // round-to-nearest-even
  union { unsigned short s; bf16_t b; } o; o.s = (unsigned short)(r >> 16);
  return o.b;
}

// pack two floats as bf16 pair (lo, hi) into one dword
__device__ inline unsigned pk2(float a, float b) {
  union { float f; unsigned u; } x, y; x.f = a; y.f = b;
  unsigned ra = (x.u + 0x7FFFu + ((x.u >> 16) & 1u)) >> 16;
  unsigned rb = (y.u + 0x7FFFu + ((y.u >> 16) & 1u)) & 0xFFFF0000u;
  return ra | rb;
}

__device__ inline void atomicMaxF(float* addr, float val) {
  int* ia = (int*)addr;
  int old = *ia;
  while (val > __int_as_float(old)) {
    int assumed = old;
    old = atomicCAS(ia, assumed, __float_as_int(val));
    if (old == assumed) break;
  }
}

__device__ inline void edge_sd(const int* __restrict__ ei, int e, int& s, int& d) {
  if (e < NE) { s = ei[e]; d = ei[NE + e]; }
  else        { s = e - NE; d = e - NE; }   // self loops appended
}

// ---- wave-level 32x32 WMMA step over one K=32 LDS stage --------------------
__device__ inline void wmma_step(const bf16_t* __restrict__ Alds,
                                 const bf16_t* __restrict__ Blds,
                                 int wm, int wn, int lane, v8f acc[2][2]) {
  const int mr  = lane & 15;
  const int akb = (lane >> 4) << 3;   // A: K base 0 or 8 (halves at +0 / +16)
  const int bkb = (lane >> 4) << 3;   // B: K base 0 or 8 (halves at +0 / +16)
  FragU a[2], b[2];
#pragma unroll
  for (int mi = 0; mi < 2; ++mi) {
    const bf16_t* p = Alds + (wm + mi * 16 + mr) * LDSS + akb;
    a[mi].q[0] = *(const U128*)p;
    a[mi].q[1] = *(const U128*)(p + 16);
  }
#pragma unroll
  for (int ni = 0; ni < 2; ++ni) {
    const bf16_t* p = Blds + (wn + ni * 16 + mr) * LDSS + bkb;
    b[ni].q[0] = *(const U128*)p;
    b[ni].q[1] = *(const U128*)(p + 16);
  }
#pragma unroll
  for (int mi = 0; mi < 2; ++mi)
#pragma unroll
    for (int ni = 0; ni < 2; ++ni)
      acc[mi][ni] = __builtin_amdgcn_wmma_f32_16x16x32_bf16(
          false, a[mi].v, false, b[ni].v, (short)0, acc[mi][ni], false, false);
}

// ---- packed-operand GEMM: C[M,N] = act(A @ B^T + bias) ---------------------
// A: bf16 [M][Kp]  (Kp mult of 32, zero padded)
// B: bf16 [Np][Kp] (transposed weights, Np mult of 64, zero padded)
// 128x64 macro-tile, 256 threads = 8 waves (4x2 grid of 32x32 wave tiles)
__global__ __launch_bounds__(256) void gemm_bf16_kernel(
    const bf16_t* __restrict__ A, const bf16_t* __restrict__ B,
    const float* __restrict__ bias, float* __restrict__ C,
    int M, int N, int Kp, int ldc, int relu) {
  __shared__ bf16_t __attribute__((aligned(16))) Alds[128 * LDSS];
  __shared__ bf16_t __attribute__((aligned(16))) Blds[64 * LDSS];
  const int tid  = threadIdx.x;
  const int lane = tid & 31;
  const int w    = tid >> 5;
  const int m0   = blockIdx.y * 128;
  const int n0   = blockIdx.x * 64;
  const int wm   = (w >> 1) * 32;   // 0..96
  const int wn   = (w & 1) * 32;

  v8f z = {};
  v8f acc[2][2]; acc[0][0] = z; acc[0][1] = z; acc[1][0] = z; acc[1][1] = z;

  for (int k0 = 0; k0 < Kp; k0 += 32) {
    // A tile 128x32: 512 octets, 2 passes, guard-free (row clamped)
#pragma unroll
    for (int j = 0; j < 2; ++j) {
      int grp = tid + 256 * j;
      int r = grp >> 2, cg = (grp & 3) << 3;
      int gm = m0 + r; gm = gm < M ? gm : M - 1;
      U128 p = *(const U128*)(A + (size_t)gm * Kp + k0 + cg);
      *(U128*)(Alds + r * LDSS + cg) = p;
    }
    // B tile 64x32: 256 octets, 1 pass, guard-free (padded)
    {
      int r = tid >> 2, cg = (tid & 3) << 3;
      U128 p = *(const U128*)(B + (size_t)(n0 + r) * Kp + k0 + cg);
      *(U128*)(Blds + r * LDSS + cg) = p;
    }
    __syncthreads();
    wmma_step(Alds, Blds, wm, wn, lane, acc);
    __syncthreads();
  }

  const int rowoff = (lane >> 4) << 3;
  const int coloff = lane & 15;
#pragma unroll
  for (int mi = 0; mi < 2; ++mi)
#pragma unroll
    for (int ni = 0; ni < 2; ++ni)
#pragma unroll
      for (int r = 0; r < 8; ++r) {
        int gm = m0 + wm + mi * 16 + r + rowoff;
        int gn = n0 + wn + ni * 16 + coloff;
        if (gm < M && gn < N) {
          float v = acc[mi][ni][r];
          if (bias) v += bias[gn];
          if (relu) v = v > 0.0f ? v : 0.0f;
          C[(size_t)gm * ldc + gn] = v;
        }
      }
}

// ---- conv as batched GEMM: Q[g] = S_g^T[128x1024] @ Wp[1024x256] -----------
// embB: bf16 [VOC+1][EMBD], row VOC == zeros; tpad: [NG][SEQP] padded tokens
__global__ __launch_bounds__(256) void conv_gemm_kernel(
    const unsigned short* __restrict__ embB, const int* __restrict__ tpad,
    const bf16_t* __restrict__ Wpt,   // [256][WPLD] K-major, zero padded
    float* __restrict__ Q) {
  __shared__ bf16_t __attribute__((aligned(16))) Alds[128 * LDSS];
  __shared__ bf16_t __attribute__((aligned(16))) Blds[64 * LDSS];
  const int tid  = threadIdx.x;
  const int lane = tid & 31;
  const int w    = tid >> 5;
  const int g    = blockIdx.z;
  const int n0   = blockIdx.x * 64;   // col (k*32+f) in [0,256)
  const int wm   = (w >> 1) * 32;
  const int wn   = (w & 1) * 32;

  v8f z = {};
  v8f acc[2][2]; acc[0][0] = z; acc[0][1] = z; acc[1][0] = z; acc[1][1] = z;

  for (int k0 = 0; k0 < SEQP; k0 += 32) {
    // A: gathered embedding columns; A[u,i] = embB[tpad[g,i]*128 + u]; guard-free
#pragma unroll
    for (int j = 0; j < 4; ++j) {
      int grp = tid + 256 * j;
      int r = grp >> 3, cg = (grp & 7) << 2;   // r = u (0..127), cg = i base
      const int* tp = tpad + (size_t)g * SEQP + k0 + cg;
      unsigned u0 = embB[(size_t)tp[0] * EMBD + r];
      unsigned u1 = embB[(size_t)tp[1] * EMBD + r];
      unsigned u2 = embB[(size_t)tp[2] * EMBD + r];
      unsigned u3 = embB[(size_t)tp[3] * EMBD + r];
      U64 p; p.x = u0 | (u1 << 16); p.y = u2 | (u3 << 16);
      *(U64*)(Alds + r * LDSS + cg) = p;
    }
    // B: guard-free packed copy from padded transposed weights
    {
      int r = tid >> 2, cg = (tid & 3) << 3;
      U128 p = *(const U128*)(Wpt + (size_t)(n0 + r) * WPLD + k0 + cg);
      *(U128*)(Blds + r * LDSS + cg) = p;
    }
    __syncthreads();
    wmma_step(Alds, Blds, wm, wn, lane, acc);
    __syncthreads();
  }

  const int rowoff = (lane >> 4) << 3;
  const int coloff = lane & 15;
  float* Qg = Q + (size_t)g * (EMBD * 256);
#pragma unroll
  for (int mi = 0; mi < 2; ++mi)
#pragma unroll
    for (int ni = 0; ni < 2; ++ni)
#pragma unroll
      for (int r = 0; r < 8; ++r) {
        int gm = wm + mi * 16 + r + rowoff;
        int gn = n0 + wn + ni * 16 + coloff;
        Qg[gm * 256 + gn] = acc[mi][ni][r];
      }
}

// ---- operand packing kernels ----------------------------------------------
// A pack: f32 [M,K] -> bf16 [M][Kp], zero padded, dword stores
__global__ void k_packA(const float* __restrict__ src, bf16_t* __restrict__ dst,
                        int M, int K, int Kp) {
  int idx = blockIdx.x * 256 + threadIdx.x;    // over M*(Kp/2)
  int kh = Kp >> 1;
  if (idx >= M * kh) return;
  int m = idx / kh, k = (idx - m * kh) * 2;
  float a = (k     < K) ? src[(size_t)m * K + k]     : 0.0f;
  float b = (k + 1 < K) ? src[(size_t)m * K + k + 1] : 0.0f;
  ((unsigned*)dst)[idx] = pk2(a, b);
}

// B pack (transpose): f32 [K,N] -> bf16 [Np][Kp], zero padded
__global__ void k_packB(const float* __restrict__ src, bf16_t* __restrict__ dst,
                        int K, int N, int Kp, int Np) {
  int idx = blockIdx.x * 256 + threadIdx.x;    // over Np*(Kp/2)
  int kh = Kp >> 1;
  if (idx >= Np * kh) return;
  int n = idx / kh, k = (idx - n * kh) * 2;
  float a = (n < N && k     < K) ? src[(size_t)k       * N + n] : 0.0f;
  float b = (n < N && k + 1 < K) ? src[(size_t)(k + 1) * N + n] : 0.0f;
  ((unsigned*)dst)[idx] = pk2(a, b);
}

// conv weights: Wpt[col=k*32+f][i], i padded to WPLD with zeros
__global__ void k_wprime(const float* __restrict__ Wc, bf16_t* __restrict__ Wpt) {
  int idx = blockIdx.x * 256 + threadIdx.x;      // over 256*WPLD
  if (idx >= 256 * WPLD) return;
  int col = idx / WPLD, i = idx % WPLD;
  int k = col / 32, f = col % 32;                // col = k*32+f
  float v = (i < SEQL) ? Wc[((size_t)f * SEQL + i) * 8 + k] : 0.0f;
  Wpt[idx] = f2bf(v);
}

// bf16 embedding table with sentinel zero row VOC
__global__ void k_embB(const float* __restrict__ emb, bf16_t* __restrict__ embB) {
  int idx = blockIdx.x * 256 + threadIdx.x;      // over (VOC+1)*EMBD
  if (idx >= (VOC + 1) * EMBD) return;
  int row = idx / EMBD;
  embB[idx] = f2bf(row < VOC ? emb[idx] : 0.0f);
}

// padded token stream; pad entries -> sentinel VOC
__global__ void k_tpad(const int* __restrict__ target, int* __restrict__ tpad) {
  int idx = blockIdx.x * 256 + threadIdx.x;      // over NG*SEQP
  if (idx >= NG * SEQP) return;
  int g = idx / SEQP, i = idx % SEQP;
  tpad[idx] = (i < SEQL) ? target[g * SEQL + i] : VOC;
}

// ---- small kernels ---------------------------------------------------------
__global__ void k_fill(float* p, float v, int n) {
  int i = blockIdx.x * 256 + threadIdx.x;
  if (i < n) p[i] = v;
}

__global__ void k_att(const float* __restrict__ h, const float* __restrict__ as,
                      const float* __restrict__ ad, float* __restrict__ a_s,
                      float* __restrict__ a_d) {
  int idx = blockIdx.x * 256 + threadIdx.x;
  if (idx >= NN * NH) return;
  int n = idx / NH, hh = idx % NH;
  const float* hp = h + (size_t)n * HF + hh * FX;
  const float* sp = as + hh * FX;
  const float* dp = ad + hh * FX;
  float s1 = 0.f, s2 = 0.f;
  for (int c = 0; c < FX; ++c) { float v = hp[c]; s1 += v * sp[c]; s2 += v * dp[c]; }
  a_s[idx] = s1; a_d[idx] = s2;
}

__global__ void k_edge1(const int* __restrict__ ei, const float* __restrict__ a_s,
                        const float* __restrict__ a_d, float* __restrict__ alpha,
                        float* __restrict__ mbuf, float* __restrict__ deg) {
  int idx = blockIdx.x * 256 + threadIdx.x;
  if (idx >= E2 * NH) return;
  int e = idx / NH, hh = idx % NH;
  int s, d; edge_sd(ei, e, s, d);
  float a = a_s[s * NH + hh] + a_d[d * NH + hh];
  a = a > 0.f ? a : 0.2f * a;                      // leaky_relu 0.2
  alpha[idx] = a;
  atomicMaxF(&mbuf[d * NH + hh], a);
  if (hh == 0) atomicAdd(&deg[d], 1.0f);
}

__global__ void k_edge2(const int* __restrict__ ei, float* __restrict__ alpha,
                        const float* __restrict__ mbuf, float* __restrict__ sden) {
  int idx = blockIdx.x * 256 + threadIdx.x;
  if (idx >= E2 * NH) return;
  int e = idx / NH, hh = idx % NH;
  int s, d; edge_sd(ei, e, s, d); (void)s;
  float ev = expf(alpha[idx] - mbuf[d * NH + hh]);
  alpha[idx] = ev;
  atomicAdd(&sden[d * NH + hh], ev);
}

__global__ __launch_bounds__(128) void k_edge_agg(
    const int* __restrict__ ei, const float* __restrict__ h,
    const float* __restrict__ eexp, const float* __restrict__ sden,
    float* __restrict__ agg) {
  int e = blockIdx.x;
  int s, d; edge_sd(ei, e, s, d);
  for (int hc = threadIdx.x; hc < HF; hc += 128) {
    int hh = hc / FX;
    float att = eexp[e * NH + hh] / (sden[d * NH + hh] + 1e-16f);
    atomicAdd(&agg[(size_t)d * HF + hc], h[(size_t)s * HF + hc] * att);
  }
}

__global__ void k_bias_relu(float* __restrict__ x, const float* __restrict__ b,
                            int total, int C) {
  int i = blockIdx.x * 256 + threadIdx.x;
  if (i < total) { float v = x[i] + b[i % C]; x[i] = v > 0.f ? v : 0.f; }
}

__global__ __launch_bounds__(128) void k_gcn_agg(
    const int* __restrict__ ei, const float* __restrict__ h2,
    const float* __restrict__ deg, float* __restrict__ x2) {
  int e = blockIdx.x;
  int s, d; edge_sd(ei, e, s, d);
  float nrm = rsqrtf(fmaxf(deg[s], 1.0f)) * rsqrtf(fmaxf(deg[d], 1.0f));
  for (int c = threadIdx.x; c < HF; c += 128)
    atomicAdd(&x2[(size_t)d * HF + c], h2[(size_t)s * HF + c] * nrm);
}

__global__ __launch_bounds__(128) void k_pool(
    const float* __restrict__ x2, const int* __restrict__ batch,
    float* __restrict__ ssum, float* __restrict__ cnt) {
  int n = blockIdx.x;
  int g = batch[n];
  if (threadIdx.x == 0) atomicAdd(&cnt[g], 1.0f);
  for (int c = threadIdx.x; c < HF; c += 128)
    atomicAdd(&ssum[(size_t)g * HF + c], x2[(size_t)n * HF + c]);
}

__global__ void k_gx(const float* __restrict__ ssum, const float* __restrict__ cnt,
                     float* __restrict__ gx) {
  int idx = blockIdx.x * 256 + threadIdx.x;
  if (idx >= NG * 1560) return;
  int g = idx / 1560, c = idx % 1560;
  float cn = fmaxf(cnt[g], 1.0f);
  gx[idx] = (c < HF) ? ssum[g * HF + c] / cn : ssum[g * HF + (c - HF)];
}

__global__ void k_conv_epi(const float* __restrict__ Q, const float* __restrict__ bc,
                           float* __restrict__ cflat) {
  int idx = blockIdx.x * 256 + threadIdx.x;      // over NG*CFL
  if (idx >= NG * CFL) return;
  int g = idx / CFL, rem = idx % CFL;
  int f = rem / CO, t = rem % CO;
  float s = bc[f];
#pragma unroll
  for (int k = 0; k < 8; ++k)
    s += Q[(size_t)g * (EMBD * 256) + (t + k) * 256 + (k * 32 + f)];
  cflat[idx] = s;
}

__global__ __launch_bounds__(128) void k_final(
    const float* __restrict__ h2f, const float* __restrict__ Wo,
    const float* __restrict__ bo, float* __restrict__ out) {
  __shared__ float red[128];
  int g = blockIdx.x;
  float s = 0.f;
  for (int c = threadIdx.x; c < 512; c += 128) s += h2f[g * 512 + c] * Wo[c];
  red[threadIdx.x] = s;
  __syncthreads();
  for (int st = 64; st > 0; st >>= 1) {
    if ((int)threadIdx.x < st) red[threadIdx.x] += red[threadIdx.x + st];
    __syncthreads();
  }
  if (threadIdx.x == 0) out[g] = red[0] + bo[0];
}

// ---- host launch -----------------------------------------------------------
extern "C" void kernel_launch(void* const* d_in, const int* in_sizes, int n_in,
                              void* d_out, int out_size, void* d_ws, size_t ws_size,
                              hipStream_t stream) {
  (void)in_sizes; (void)n_in; (void)out_size; (void)ws_size;
  const float* x       = (const float*)d_in[0];
  const float* W_gat   = (const float*)d_in[1];
  const float* att_src = (const float*)d_in[2];
  const float* att_dst = (const float*)d_in[3];
  const float* b_gat   = (const float*)d_in[4];
  const float* W_gcn   = (const float*)d_in[5];
  const float* b_gcn   = (const float*)d_in[6];
  const float* W_g1    = (const float*)d_in[7];
  const float* b_g1    = (const float*)d_in[8];
  const float* W_g2    = (const float*)d_in[9];
  const float* b_g2    = (const float*)d_in[10];
  const float* emb_xt  = (const float*)d_in[11];
  const float* W_conv  = (const float*)d_in[12];
  const float* b_conv  = (const float*)d_in[13];
  const float* W_xt    = (const float*)d_in[14];
  const float* b_xt    = (const float*)d_in[15];
  const float* W_1     = (const float*)d_in[16];
  const float* b_1     = (const float*)d_in[17];
  const float* W_2     = (const float*)d_in[18];
  const float* b_2     = (const float*)d_in[19];
  const float* W_out   = (const float*)d_in[20];
  const float* b_out   = (const float*)d_in[21];
  const int* edge_index = (const int*)d_in[22];
  const int* batch      = (const int*)d_in[23];
  const int* target     = (const int*)d_in[24];
  float* out = (float*)d_out;

  char* ws = (char*)d_ws;
  size_t off = 0;
  auto A_ = [&](size_t bytes) { size_t o = off; off += (bytes + 255) & ~(size_t)255; return o; };
  float*  buf1 = (float*)(ws + A_((size_t)NG * EMBD * 256 * 4)); // 67MB: h -> h2 -> Q
  float*  buf2 = (float*)(ws + A_((size_t)NN * HF * 4));         // agg/x1 -> cflat
  float*  buf3 = (float*)(ws + A_((size_t)NN * HF * 4));         // x2
  bf16_t* Apk  = (bf16_t*)(ws + A_((size_t)NN * 800 * 2));       // packed A (reused)
  bf16_t* Bpk  = (bf16_t*)(ws + A_((size_t)1536 * 1568 * 2));    // packed B^T (reused)
  float*  a_s  = (float*)(ws + A_((size_t)NN * NH * 4));
  float*  a_d  = (float*)(ws + A_((size_t)NN * NH * 4));
  float*  mbuf = (float*)(ws + A_((size_t)NN * NH * 4));
  float*  sden = (float*)(ws + A_((size_t)NN * NH * 4));
  float*  eexp = (float*)(ws + A_((size_t)E2 * NH * 4));
  float*  deg  = (float*)(ws + A_((size_t)NN * 4));
  float*  ssum = (float*)(ws + A_((size_t)NG * HF * 4));
  float*  cnt  = (float*)(ws + A_((size_t)NG * 4));
  float*  gx   = (float*)(ws + A_((size_t)NG * 1560 * 4));
  float*  g1o  = (float*)(ws + A_((size_t)NG * 1500 * 4));
  float*  xc   = (float*)(ws + A_((size_t)NG * 256 * 4));
  float*  h1b  = (float*)(ws + A_((size_t)NG * 1024 * 4));
  float*  h2f  = (float*)(ws + A_((size_t)NG * 512 * 4));
  bf16_t* Wpt  = (bf16_t*)(ws + A_((size_t)256 * WPLD * 2));
  bf16_t* embB = (bf16_t*)(ws + A_((size_t)(VOC + 1) * EMBD * 2));
  int*    tpad = (int*)(ws + A_((size_t)NG * SEQP * 4));

  float* hmat  = buf1;   // [NN, HF]
  float* aggb  = buf2;   // agg -> x1 (in place)
  float* h2m   = buf1;   // reuse after h dead
  float* x2m   = buf3;
  float* Q     = buf1;   // reuse after h2 dead
  float* cflat = buf2;   // reuse after x1 dead

  auto gemm = [&](const float* Af, const float* Bf, const float* bias, float* C,
                  int M, int N, int K, int ldc, int relu) {
    int Kp = (K + 31) & ~31;
    int Np = (N + 63) & ~63;
    int ta = M * (Kp >> 1), tb = Np * (Kp >> 1);
    k_packA<<<(ta + 255) / 256, 256, 0, stream>>>(Af, Apk, M, K, Kp);
    k_packB<<<(tb + 255) / 256, 256, 0, stream>>>(Bf, Bpk, K, N, Kp, Np);
    dim3 g(Np / 64, (M + 127) / 128);
    gemm_bf16_kernel<<<g, dim3(256), 0, stream>>>(Apk, Bpk, bias, C, M, N, Kp, ldc, relu);
  };

  // init accumulators
  hipMemsetAsync(aggb, 0, (size_t)NN * HF * 4, stream);
  hipMemsetAsync(x2m,  0, (size_t)NN * HF * 4, stream);
  hipMemsetAsync(sden, 0, (size_t)NN * NH * 4, stream);
  hipMemsetAsync(deg,  0, (size_t)NN * 4, stream);
  hipMemsetAsync(ssum, 0, (size_t)NG * HF * 4, stream);
  hipMemsetAsync(cnt,  0, (size_t)NG * 4, stream);
  k_fill<<<(NN * NH + 255) / 256, 256, 0, stream>>>(mbuf, -1e30f, NN * NH);

  // ---- GAT branch ----
  gemm(x, W_gat, nullptr, hmat, NN, HF, FX, HF, 0);                 // h = x@W_gat
  k_att<<<(NN * NH + 255) / 256, 256, 0, stream>>>(hmat, att_src, att_dst, a_s, a_d);
  k_edge1<<<(E2 * NH + 255) / 256, 256, 0, stream>>>(edge_index, a_s, a_d, eexp, mbuf, deg);
  k_edge2<<<(E2 * NH + 255) / 256, 256, 0, stream>>>(edge_index, eexp, mbuf, sden);
  k_edge_agg<<<E2, 128, 0, stream>>>(edge_index, hmat, eexp, sden, aggb);
  k_bias_relu<<<(NN * HF + 255) / 256, 256, 0, stream>>>(aggb, b_gat, NN * HF, HF);

  // ---- GCN ----
  gemm(aggb, W_gcn, nullptr, h2m, NN, HF, HF, HF, 0);               // h2 = x1@W_gcn
  k_gcn_agg<<<E2, 128, 0, stream>>>(edge_index, h2m, deg, x2m);
  k_bias_relu<<<(NN * HF + 255) / 256, 256, 0, stream>>>(x2m, b_gcn, NN * HF, HF);

  // ---- pooling + graph MLP ----
  k_pool<<<NN, 128, 0, stream>>>(x2m, batch, ssum, cnt);
  k_gx<<<(NG * 1560 + 255) / 256, 256, 0, stream>>>(ssum, cnt, gx);
  gemm(gx, W_g1, b_g1, g1o, NG, 1500, 1560, 1500, 1);
  gemm(g1o, W_g2, b_g2, xc, NG, 128, 1500, 256, 0);                 // xc[:, :128]

  // ---- protein conv branch ----
  k_wprime<<<(256 * WPLD + 255) / 256, 256, 0, stream>>>(W_conv, Wpt);
  k_embB<<<((VOC + 1) * EMBD + 255) / 256, 256, 0, stream>>>(emb_xt, embB);
  k_tpad<<<(NG * SEQP + 255) / 256, 256, 0, stream>>>(target, tpad);
  conv_gemm_kernel<<<dim3(4, 1, NG), 256, 0, stream>>>(
      (const unsigned short*)embB, tpad, Wpt, Q);
  k_conv_epi<<<(NG * CFL + 255) / 256, 256, 0, stream>>>(Q, b_conv, cflat);
  gemm(cflat, W_xt, b_xt, xc + 128, NG, 128, CFL, 256, 0);          // xc[:, 128:]

  // ---- fusion MLP ----
  gemm(xc, W_1, b_1, h1b, NG, 1024, 256, 1024, 1);
  gemm(h1b, W_2, b_2, h2f, NG, 512, 1024, 512, 1);
  k_final<<<NG, 128, 0, stream>>>(h2f, W_out, b_out, out);
}
